// TodTransformerEncoder_32615981646426
// MI455X (gfx1250) — compile-verified, matrix operand order checked by reading, not run on hardware
//
#include <hip/hip_runtime.h>
#include <hip/hip_bf16.h>
#include <math.h>

// ---- problem constants (from reference) ----
#define NLAYERS 6
#define NHEADS  8
#define NPOINTS 4
#define DMODEL  256
#define DFFN    1024
#define BATCH   4
#define NFOCUS  2048
#define NTOK    21760            // 128^2+64^2+32^2+16^2
#define NLEV    4

typedef __attribute__((ext_vector_type(16))) _Float16 v16h;
typedef __attribute__((ext_vector_type(8)))  float    v8f;
typedef unsigned int u32;

__device__ __forceinline__ float warp_sum(float v) {
#pragma unroll
  for (int m = 16; m >= 1; m >>= 1) v += __shfl_xor(v, m, 32);
  return v;
}

// ======================================================================
// Generic tiled GEMM: C[M,N] = act(A[M,K] @ B[K,N] + bias) , f32 in/out,
// f16 WMMA compute (v_wmma_f32_16x16x32_f16). Block tile 128x64, 8 waves,
// each wave owns a 32x32 quadrant (2x2 WMMA tiles), K-step 32.
// Staging: per-thread contiguous float4 global loads, packed-f16 b128 LDS
// stores; global_prefetch of next A slice.
// act: 0 none, 1 relu, 2 gelu(exact). rowmask!=0 -> zero masked rows.
// ======================================================================
__global__ __launch_bounds__(256)
void gemm_wmma_kernel(const float* __restrict__ A, const float* __restrict__ B,
                      const float* __restrict__ bias, float* __restrict__ C,
                      int M, int K, int N, int act,
                      const unsigned char* __restrict__ rowmask)
{
  __shared__ __align__(16) _Float16 sA[128 * 32];   // [m][k]
  __shared__ __align__(16) _Float16 sB[64 * 32];    // [n][k] (transposed)
  const int t    = threadIdx.x;
  const int m0   = blockIdx.x * 128;
  const int n0   = blockIdx.y * 64;
  const int wave = t >> 5, lane = t & 31;
  const int wr   = wave >> 1;     // 0..3 : 32-row group
  const int wc   = wave & 1;      // 0..1 : 32-col group
  const int r    = lane & 15, hh = lane >> 4;

  // A staging geometry: thread owns row (t>>1), k-half (t&1)*16
  const int am = t >> 1, ah = (t & 1) * 16;
  // B staging geometry: thread owns col (t&63), k-octet (t>>6)*8
  const int bn = t & 63, bk8 = (t >> 6) * 8;

  v8f acc[2][2];
#pragma unroll
  for (int i = 0; i < 2; ++i)
#pragma unroll
    for (int j = 0; j < 2; ++j)
#pragma unroll
      for (int e = 0; e < 8; ++e) acc[i][j][e] = 0.0f;

  for (int k0 = 0; k0 < K; k0 += 32) {
    // ---- stage A tile 128x32 (f32 -> f16), b128 in / b128 out ----
    {
      const float4* src = (const float4*)(A + (size_t)(m0 + am) * K + k0 + ah);
      union { _Float16 h[16]; uint4 q[2]; } pk;
#pragma unroll
      for (int i = 0; i < 4; ++i) {
        float4 f = src[i];
        pk.h[i * 4 + 0] = (_Float16)f.x;
        pk.h[i * 4 + 1] = (_Float16)f.y;
        pk.h[i * 4 + 2] = (_Float16)f.z;
        pk.h[i * 4 + 3] = (_Float16)f.w;
      }
      uint4* dst = (uint4*)(sA + t * 16);
      dst[0] = pk.q[0];
      dst[1] = pk.q[1];
      if (k0 + 32 < K)   // prefetch next A slice (global_prefetch_b8 path)
        __builtin_prefetch(A + (size_t)(m0 + am) * K + k0 + 32 + ah, 0, 1);
    }
    // ---- stage B tile 32x64 transposed -> sB[n][k], coalesced loads ----
    {
      union { _Float16 h[8]; uint4 q; } pk;
#pragma unroll
      for (int j = 0; j < 8; ++j)
        pk.h[j] = (_Float16)B[(size_t)(k0 + bk8 + j) * N + n0 + bn];
      *(uint4*)(sB + bn * 32 + bk8) = pk.q;
    }
    __syncthreads();

    union U { v16h v; u32 u[8]; };
    U a0, a1, b0, b1;
    const u32* pA = (const u32*)sA;
    const u32* pB = (const u32*)sB;
#pragma unroll
    for (int p = 0; p < 8; ++p) {
      // ISA 16-bit A layout: VGPR p holds K pair starting at:
      int kk = (p < 4) ? (hh * 8 + 2 * p) : (16 + hh * 8 + 2 * (p - 4));
      int kw = kk >> 1;
      a0.u[p] = pA[(wr * 32 + r)      * 16 + kw];
      a1.u[p] = pA[(wr * 32 + 16 + r) * 16 + kw];
      b0.u[p] = pB[(wc * 32 + r)      * 16 + kw];
      b1.u[p] = pB[(wc * 32 + 16 + r) * 16 + kw];
    }
    acc[0][0] = __builtin_amdgcn_wmma_f32_16x16x32_f16(false, a0.v, false, b0.v, (short)0, acc[0][0], false, false);
    acc[0][1] = __builtin_amdgcn_wmma_f32_16x16x32_f16(false, a0.v, false, b1.v, (short)0, acc[0][1], false, false);
    acc[1][0] = __builtin_amdgcn_wmma_f32_16x16x32_f16(false, a1.v, false, b0.v, (short)0, acc[1][0], false, false);
    acc[1][1] = __builtin_amdgcn_wmma_f32_16x16x32_f16(false, a1.v, false, b1.v, (short)0, acc[1][1], false, false);
    __syncthreads();
  }

  // epilogue: D layout -> row = e + 8*hh, col = r within 16x16 tile
#pragma unroll
  for (int i = 0; i < 2; ++i)
#pragma unroll
    for (int j = 0; j < 2; ++j)
#pragma unroll
      for (int e = 0; e < 8; ++e) {
        int mrow = m0 + wr * 32 + i * 16 + hh * 8 + e;
        int ncol = n0 + wc * 32 + j * 16 + r;
        float x = acc[i][j][e];
        if (bias) x += bias[ncol];
        if (act == 1)      x = fmaxf(x, 0.0f);
        else if (act == 2) x = 0.5f * x * (1.0f + erff(x * 0.70710678118654752f));
        if (rowmask && rowmask[mrow]) x = 0.0f;
        C[(size_t)mrow * N + ncol] = x;
      }
}

// ---------------- reference points (B,N,L,2) ----------------
__global__ __launch_bounds__(256)
void ref_kernel(const float* __restrict__ vr, float* __restrict__ ref)
{
  int tid = blockIdx.x * 256 + threadIdx.x;
  if (tid >= BATCH * NTOK) return;
  int b = tid / NTOK, n = tid % NTOK;
  int lvl, dim, s;
  if      (n < 16384) { lvl = 0; dim = 128; s = 0;     }
  else if (n < 20480) { lvl = 1; dim = 64;  s = 16384; }
  else if (n < 21504) { lvl = 2; dim = 32;  s = 20480; }
  else                { lvl = 3; dim = 16;  s = 21504; }
  int local = n - s;
  int gy = local / dim, gx = local % dim;
  float fx = (gx + 0.5f) / (vr[(b * NLEV + lvl) * 2 + 0] * dim);
  float fy = (gy + 0.5f) / (vr[(b * NLEV + lvl) * 2 + 1] * dim);
  float* o = ref + (size_t)tid * 8;
#pragma unroll
  for (int l = 0; l < 4; ++l) {
    o[l * 2 + 0] = fx * vr[(b * NLEV + l) * 2 + 0];
    o[l * 2 + 1] = fy * vr[(b * NLEV + l) * 2 + 1];
  }
}

// ------------- gather q/qpos/ref at foreground inds (float4) -----------
__global__ __launch_bounds__(256)
void gather_kernel(const float* __restrict__ outp, const float* __restrict__ qpos_all,
                   const float* __restrict__ reff, const int* __restrict__ inds,
                   float* __restrict__ q, float* __restrict__ qpos, float* __restrict__ qref)
{
  int tid = blockIdx.x * 256 + threadIdx.x;     // B*K*64 exact
  int bk = tid >> 6, d4 = tid & 63;
  int b  = bk >> 11;                             // K = 2048
  int idx = inds[bk];
  size_t srow = ((size_t)b * NTOK + idx) * 256;
  size_t drow = (size_t)bk * 256;
  ((float4*)(q    + drow))[d4] = ((const float4*)(outp     + srow))[d4];
  ((float4*)(qpos + drow))[d4] = ((const float4*)(qpos_all + srow))[d4];
  if (d4 < 2)
    ((float4*)(qref + (size_t)bk * 8))[d4] =
        ((const float4*)(reff + ((size_t)b * NTOK + idx) * 8))[d4];
}

__global__ __launch_bounds__(256)
void add_kernel(const float* __restrict__ a, const float* __restrict__ b,
                float* __restrict__ c, int n4)
{
  int i = blockIdx.x * 256 + threadIdx.x;
  if (i >= n4) return;
  float4 x = ((const float4*)a)[i], y = ((const float4*)b)[i];
  float4 z = { x.x + y.x, x.y + y.y, x.z + y.z, x.w + y.w };
  ((float4*)c)[i] = z;
}

// ---------------- softmax over 16 (per b,k,head) ----------------
__global__ __launch_bounds__(256)
void softmax16_kernel(float* __restrict__ aw, int ngroups)
{
  int g = blockIdx.x * 256 + threadIdx.x;
  if (g >= ngroups) return;
  float4* p4 = (float4*)(aw + (size_t)g * 16);
  float e[16];
  float4 r0 = p4[0], r1 = p4[1], r2 = p4[2], r3 = p4[3];
  e[0]=r0.x; e[1]=r0.y; e[2]=r0.z; e[3]=r0.w;
  e[4]=r1.x; e[5]=r1.y; e[6]=r1.z; e[7]=r1.w;
  e[8]=r2.x; e[9]=r2.y; e[10]=r2.z; e[11]=r2.w;
  e[12]=r3.x; e[13]=r3.y; e[14]=r3.z; e[15]=r3.w;
  float m = e[0];
#pragma unroll
  for (int i = 1; i < 16; ++i) m = fmaxf(m, e[i]);
  float s = 0.0f;
#pragma unroll
  for (int i = 0; i < 16; ++i) { e[i] = expf(e[i] - m); s += e[i]; }
  float inv = 1.0f / s;
#pragma unroll
  for (int i = 0; i < 16; ++i) e[i] *= inv;
  p4[0] = make_float4(e[0], e[1], e[2], e[3]);
  p4[1] = make_float4(e[4], e[5], e[6], e[7]);
  p4[2] = make_float4(e[8], e[9], e[10], e[11]);
  p4[3] = make_float4(e[12], e[13], e[14], e[15]);
}

// ---------------- deformable bilinear sampling ----------------
// one wave per (b,k,head); 32 lanes = Dh channels -> coalesced gathers
__global__ __launch_bounds__(256)
void sample_kernel(const float* __restrict__ v, const float* __restrict__ off,
                   const float* __restrict__ aw, const float* __restrict__ qref,
                   float* __restrict__ samp)
{
  int gw   = blockIdx.x * 8 + (threadIdx.x >> 5);   // B*K*H waves
  int lane = threadIdx.x & 31;
  int h = gw & 7, bk = gw >> 3;
  int b = bk >> 11;

  // preload broadcast operands (contiguous per head) as float4
  float offv[32], awv[16], refv[8];
  {
    const float4* po = (const float4*)(off + (size_t)bk * 256 + h * 32);
    const float4* pa = (const float4*)(aw + (size_t)bk * 128 + h * 16);
    const float4* pr = (const float4*)(qref + (size_t)bk * 8);
#pragma unroll
    for (int i = 0; i < 8; ++i) ((float4*)offv)[i] = po[i];
#pragma unroll
    for (int i = 0; i < 4; ++i) ((float4*)awv)[i] = pa[i];
#pragma unroll
    for (int i = 0; i < 2; ++i) ((float4*)refv)[i] = pr[i];
  }

  const int dims[4]   = {128, 64, 32, 16};
  const int starts[4] = {0, 16384, 20480, 21504};
  float acc = 0.0f;
#pragma unroll
  for (int l = 0; l < 4; ++l) {
    int wl = dims[l], s = starts[l];
    float fw = (float)wl;
    float rx = refv[l * 2], ry = refv[l * 2 + 1];
#pragma unroll
    for (int p = 0; p < 4; ++p) {
      int c = l * 4 + p;
      float a = awv[c];
      float x = (rx + offv[2 * c]     / fw) * fw - 0.5f;
      float y = (ry + offv[2 * c + 1] / fw) * fw - 0.5f;
      float xf = floorf(x), yf = floorf(y);
      float wx = x - xf, wy = y - yf;
      int x0 = (int)xf, y0 = (int)yf;
#pragma unroll
      for (int cy = 0; cy < 2; ++cy)
#pragma unroll
        for (int cx = 0; cx < 2; ++cx) {
          int xi = x0 + cx, yi = y0 + cy;
          float wgt = (cx ? wx : 1.0f - wx) * (cy ? wy : 1.0f - wy) * a;
          if (xi >= 0 && xi < wl && yi >= 0 && yi < wl)
            acc += v[(((size_t)b * NTOK + s + yi * wl + xi) << 8) + h * 32 + lane] * wgt;
        }
    }
  }
  samp[((size_t)bk << 8) + h * 32 + lane] = acc;
}

// ------- y = LayerNorm(a + b) * g + beta, D=256, 1 wave/row, b128 IO -----
__global__ __launch_bounds__(256)
void ln_kernel(const float* __restrict__ a, const float* __restrict__ b,
               const float* __restrict__ g, const float* __restrict__ be,
               float* __restrict__ o)
{
  int row  = blockIdx.x * 8 + (threadIdx.x >> 5);
  int lane = threadIdx.x & 31;
  const float4* pa = (const float4*)(a + (size_t)row * 256) + lane * 2;
  const float4* pb = (const float4*)(b + (size_t)row * 256) + lane * 2;
  float vals[8];
  float4 a0 = pa[0], a1 = pa[1], b0 = pb[0], b1 = pb[1];
  vals[0] = a0.x + b0.x; vals[1] = a0.y + b0.y; vals[2] = a0.z + b0.z; vals[3] = a0.w + b0.w;
  vals[4] = a1.x + b1.x; vals[5] = a1.y + b1.y; vals[6] = a1.z + b1.z; vals[7] = a1.w + b1.w;
  float s = 0.0f;
#pragma unroll
  for (int e = 0; e < 8; ++e) s += vals[e];
  s = warp_sum(s);
  float mu = s * (1.0f / 256.0f);
  float s2 = 0.0f;
#pragma unroll
  for (int e = 0; e < 8; ++e) { float d = vals[e] - mu; s2 += d * d; }
  s2 = warp_sum(s2);
  float rs = rsqrtf(s2 * (1.0f / 256.0f) + 1e-5f);
  const float4* pg = (const float4*)g  + lane * 2;
  const float4* pe = (const float4*)be + lane * 2;
  float4 g0 = pg[0], g1 = pg[1], e0 = pe[0], e1 = pe[1];
  float4 o0, o1;
  o0.x = (vals[0]-mu)*rs*g0.x + e0.x; o0.y = (vals[1]-mu)*rs*g0.y + e0.y;
  o0.z = (vals[2]-mu)*rs*g0.z + e0.z; o0.w = (vals[3]-mu)*rs*g0.w + e0.w;
  o1.x = (vals[4]-mu)*rs*g1.x + e1.x; o1.y = (vals[5]-mu)*rs*g1.y + e1.y;
  o1.z = (vals[6]-mu)*rs*g1.z + e1.z; o1.w = (vals[7]-mu)*rs*g1.w + e1.w;
  float4* po = (float4*)(o + (size_t)row * 256) + lane * 2;
  po[0] = o0; po[1] = o1;
}

// ---------------- noise head: sigma + log-sigma ----------------
__global__ __launch_bounds__(256)
void sigma_kernel(const float* __restrict__ h2, const float* __restrict__ w3,
                  const float* __restrict__ b3, float* __restrict__ sigma,
                  float* __restrict__ logb)
{
  int bk = blockIdx.x * 256 + threadIdx.x;   // B*K exact
  float s = b3[0];
  const float4* p = (const float4*)(h2 + (size_t)bk * 64);
  const float4* w = (const float4*)w3;
#pragma unroll
  for (int i = 0; i < 16; ++i) {
    float4 x = p[i], y = w[i];
    s += x.x * y.x + x.y * y.y + x.z * y.z + x.w * y.w;
  }
  float sg  = 1.0f / (1.0f + expf(-s));
  float sig = fmaxf(sg * 0.2f, 0.001f);
  sigma[bk] = sig;
  logb[bk]  = logf(sig);
}

__global__ __launch_bounds__(256)
void noise_apply_kernel(const float* __restrict__ q, const float* __restrict__ sigma,
                        const float* __restrict__ noise, float* __restrict__ nq, int n4)
{
  int i = blockIdx.x * 256 + threadIdx.x;
  if (i >= n4) return;
  float sg = sigma[i >> 6];
  float4 x = ((const float4*)q)[i], z = ((const float4*)noise)[i];
  float4 r = { x.x + sg * z.x, x.y + sg * z.y, x.z + sg * z.z, x.w + sg * z.w };
  ((float4*)nq)[i] = r;
}

// deterministic single-block reduction for the penalty scalar
__global__ __launch_bounds__(256)
void penalty_kernel(const float* __restrict__ logb, float* __restrict__ out, int n)
{
  __shared__ float sm[256];
  float s = 0.0f;
  for (int i = threadIdx.x; i < n; i += 256) s += logb[i];
  sm[threadIdx.x] = s;
  __syncthreads();
  for (int st = 128; st > 0; st >>= 1) {
    if (threadIdx.x < st) sm[threadIdx.x] += sm[threadIdx.x + st];
    __syncthreads();
  }
  if (threadIdx.x == 0) out[0] = sm[0] / (float)n + 1.4189385332046727f; // 0.5*ln(2pi)+0.5
}

// ---------------- scatter q back into output at inds (masked) ----------
__global__ __launch_bounds__(256)
void scatter_kernel(const float* __restrict__ q, const int* __restrict__ inds,
                    const int* __restrict__ fnum, float* __restrict__ outp)
{
  int tid = blockIdx.x * 256 + threadIdx.x;   // B*K*64 exact
  int bk = tid >> 6, d4 = tid & 63;
  int b = bk >> 11, k = bk & 2047;
  if (k < fnum[b]) {
    int idx = inds[bk];
    ((float4*)(outp + ((size_t)b * NTOK + idx) * 256))[d4] =
        ((const float4*)(q + (size_t)bk * 256))[d4];
  }
}

// ======================================================================
extern "C" void kernel_launch(void* const* d_in, const int* in_sizes, int n_in,
                              void* d_out, int out_size, void* d_ws, size_t ws_size,
                              hipStream_t stream)
{
  (void)in_sizes; (void)n_in; (void)out_size; (void)ws_size;

  // inputs flattened in setup_inputs() dict order (params depth-first)
  const float* query      = (const float*)d_in[0];
  const float* qpos_all   = (const float*)d_in[1];
  const float* vr         = (const float*)d_in[2];
  const float* unit_noise = (const float*)d_in[3];
  const float* W_off  = (const float*)d_in[4];
  const float* b_off  = (const float*)d_in[5];
  const float* W_attn = (const float*)d_in[6];
  const float* b_attn = (const float*)d_in[7];
  const float* W_v    = (const float*)d_in[8];
  const float* b_v    = (const float*)d_in[9];
  const float* W_o    = (const float*)d_in[10];
  const float* b_o    = (const float*)d_in[11];
  const float* ln1_g  = (const float*)d_in[12];
  const float* ln1_b  = (const float*)d_in[13];
  const float* W1     = (const float*)d_in[14];
  const float* b1     = (const float*)d_in[15];
  const float* W2     = (const float*)d_in[16];
  const float* b2     = (const float*)d_in[17];
  const float* ln2_g  = (const float*)d_in[18];
  const float* ln2_b  = (const float*)d_in[19];
  const float* nW1 = (const float*)d_in[20];
  const float* nb1 = (const float*)d_in[21];
  const float* nW2 = (const float*)d_in[22];
  const float* nb2 = (const float*)d_in[23];
  const float* nW3 = (const float*)d_in[24];
  const float* nb3 = (const float*)d_in[25];
  // d_in[26]/[27]: spatial_shapes / level_start_index (int64) -> compile-time constants
  const int* fg   = (const int*)d_in[28];
  const int* fnum = (const int*)d_in[29];
  const unsigned char* pad = (const unsigned char*)d_in[30];

  const size_t BND = (size_t)BATCH * NTOK * DMODEL;    // 22,282,240
  const size_t BKD = (size_t)BATCH * NFOCUS * DMODEL;  //  2,097,152
  const int    BK  = BATCH * NFOCUS;                   //  8192

  // outputs: (output, q, noise_query, penalty, sigma)
  float* out_output  = (float*)d_out;
  float* out_q       = out_output + BND;
  float* out_nq      = out_q + BKD;
  float* out_penalty = out_nq + BKD;
  float* out_sigma   = out_penalty + 1;

  // workspace layout (floats)
  float* ws = (float*)d_ws;
  const size_t REF  = 0;                                // B*N*8
  const size_t V    = REF  + (size_t)BATCH * NTOK * 8;  // B*N*256
  const size_t QPOS = V    + BND;                       // B*K*256
  const size_t QREF = QPOS + BKD;                       // B*K*8
  const size_t QIN  = QREF + (size_t)BK * 8;            // B*K*256
  const size_t OFF  = QIN  + BKD;                       // B*K*256
  const size_t AW   = OFF  + BKD;                       // B*K*128
  const size_t SAMP = AW   + (size_t)BK * 128;          // B*K*256
  const size_t XBUF = SAMP + BKD;                       // B*K*256
  const size_t FFN  = XBUF + BKD;                       // B*K*1024
  const size_t H1   = FFN  + (size_t)BK * 1024;         // B*K*128
  const size_t H2   = H1   + (size_t)BK * 128;          // B*K*64
  const size_t LOGB = H2   + (size_t)BK * 64;           // B*K

  auto gemm = [&](const float* A, const float* Bm, const float* bias, float* C,
                  int M, int K, int N, int act, const unsigned char* rm) {
    dim3 grid(M / 128, N / 64);
    gemm_wmma_kernel<<<grid, 256, 0, stream>>>(A, Bm, bias, C, M, K, N, act, rm);
  };

  // output = query
  hipMemcpyAsync(out_output, query, BND * sizeof(float), hipMemcpyDeviceToDevice, stream);
  // reference points
  ref_kernel<<<(BATCH * NTOK + 255) / 256, 256, 0, stream>>>(vr, ws + REF);

  for (int lid = 0; lid < NLAYERS; ++lid) {
    const int* inds = fg + (size_t)lid * BK;

    gather_kernel<<<(unsigned)(BK * 64 / 256), 256, 0, stream>>>(
        out_output, qpos_all, ws + REF, inds, out_q, ws + QPOS, ws + QREF);

    if (lid == 0) {
      // noise MLP on q1 = gathered query: 256 -gelu-> 128 -gelu-> 64 -> sigma
      gemm(out_q,   nW1, nb1, ws + H1, BK, 256, 128, 2, nullptr);
      gemm(ws + H1, nW2, nb2, ws + H2, BK, 128, 64,  2, nullptr);
      sigma_kernel<<<BK / 256, 256, 0, stream>>>(ws + H2, nW3, nb3, out_sigma, ws + LOGB);
      penalty_kernel<<<1, 256, 0, stream>>>(ws + LOGB, out_penalty, BK);
      noise_apply_kernel<<<(unsigned)(BKD / 4 / 256), 256, 0, stream>>>(
          out_q, out_sigma, unit_noise, out_nq, (int)(BKD / 4));
    }

    // value projection: v = query @ W_v + b_v  (masked by pad_mask rows)
    gemm(query, W_v + (size_t)lid * 65536, b_v + lid * 256, ws + V,
         BATCH * NTOK, 256, 256, 0, pad);

    float* branch_q[2] = { out_q, out_nq };
    for (int br = 0; br < 2; ++br) {
      float* qb = branch_q[br];
      // qin = q + qpos
      add_kernel<<<(unsigned)(BKD / 4 / 256), 256, 0, stream>>>(
          qb, ws + QPOS, ws + QIN, (int)(BKD / 4));
      // sampling offsets + attention weights
      gemm(ws + QIN, W_off  + (size_t)lid * 65536, b_off  + lid * 256, ws + OFF, BK, 256, 256, 0, nullptr);
      gemm(ws + QIN, W_attn + (size_t)lid * 32768, b_attn + lid * 128, ws + AW,  BK, 256, 128, 0, nullptr);
      softmax16_kernel<<<(BK * NHEADS) / 256, 256, 0, stream>>>(ws + AW, BK * NHEADS);
      // deformable sampling (einsum over levels/points folded in)
      sample_kernel<<<(BK * NHEADS) / 8, 256, 0, stream>>>(
          ws + V, ws + OFF, ws + AW, ws + QREF, ws + SAMP);
      // output projection, residual + LN1
      gemm(ws + SAMP, W_o + (size_t)lid * 65536, b_o + lid * 256, ws + FFN, BK, 256, 256, 0, nullptr);
      ln_kernel<<<BK / 8, 256, 0, stream>>>(qb, ws + FFN, ln1_g + lid * 256, ln1_b + lid * 256, ws + XBUF);
      // FFN: 256 -relu-> 1024 -> 256, residual + LN2 -> back into branch buffer
      gemm(ws + XBUF, W1 + (size_t)lid * 262144, b1 + lid * 1024, ws + FFN, BK, 256, 1024, 1, nullptr);
      gemm(ws + FFN,  W2 + (size_t)lid * 262144, b2 + lid * 256,  ws + OFF, BK, 1024, 256, 0, nullptr);
      ln_kernel<<<BK / 8, 256, 0, stream>>>(ws + XBUF, ws + OFF, ln2_g + lid * 256, ln2_b + lid * 256, qb);
    }

    // scatter q back into output (only first focus_token_nums[b] tokens)
    scatter_kernel<<<(unsigned)(BK * 64 / 256), 256, 0, stream>>>(out_q, inds, fnum, out_output);
  }
}
